// AxialRotaryMHA_41506563948998
// MI455X (gfx1250) — matrix operand order repaired and checked by reading, not compiled, not run on hardware
//
#include <hip/hip_runtime.h>

typedef __attribute__((ext_vector_type(16))) _Float16 v16h;
typedef __attribute__((ext_vector_type(8)))  _Float16 v8h;
typedef __attribute__((ext_vector_type(8)))  float    v8f;

#define D_EMB 512
#define SEQ   4096
#define NB    2
#define NH    8
#define HD    64
#define WELEM (D_EMB * D_EMB)           // 262144
#define XELEM ((size_t)NB * SEQ * D_EMB) // 4194304

// ---- fragment loaders (CDNA5 wave32 WMMA layouts, ISA 7.12.2) ----

// A-matrix 16x32 f16: lane 0-15 = row lane, K chunks [0..7],[16..23];
// lanes 16-31 = row lane-16, K chunks [8..15],[24..31].
__device__ __forceinline__ v16h load_frag_a_h(const _Float16* base, int stride, int lane) {
  int row = lane & 15;
  int k0  = (lane >> 4) << 3;            // 0 or 8
  const _Float16* p = base + row * stride + k0;
  v8h lo = *(const v8h*)(p);
  v8h hi = *(const v8h*)(p + 16);
  return __builtin_shufflevector(lo, hi, 0,1,2,3,4,5,6,7,8,9,10,11,12,13,14,15);
}

// B-matrix 32x16 f16: lane 0-15 = col lane, K chunks [0..7],[16..23];
// lanes 16-31 = col lane-16, K chunks [8..15],[24..31]. Source row-major [n][k].
__device__ __forceinline__ v16h load_frag_b_h(const _Float16* base, int stride, int lane) {
  int n  = lane & 15;
  int k0 = (lane >> 4) << 3;             // 0 or 8
  const _Float16* p = base + (size_t)n * stride + k0;
  v8h lo = *(const v8h*)(p);
  v8h hi = *(const v8h*)(p + 16);
  return __builtin_shufflevector(lo, hi, 0,1,2,3,4,5,6,7,8,9,10,11,12,13,14,15);
}

__device__ __forceinline__ v8f wmma_f16(v16h a, v16h b, v8f c) {
  return __builtin_amdgcn_wmma_f32_16x16x32_f16(false, a, false, b, (short)0, c, false, false);
}

// ---- Kernel 0: bulk f32 -> f16 conversion (weights + x), one-time pre-pass ----
__global__ __launch_bounds__(256)
void cvt_f32_f16_kernel(const float* __restrict__ src, _Float16* __restrict__ dst) {
  size_t i = ((size_t)blockIdx.x * 256 + threadIdx.x) * 8;
  float4 a = *(const float4*)(src + i);
  float4 b = *(const float4*)(src + i + 4);
  v8h r;
  r[0]=(_Float16)a.x; r[1]=(_Float16)a.y; r[2]=(_Float16)a.z; r[3]=(_Float16)a.w;
  r[4]=(_Float16)b.x; r[5]=(_Float16)b.y; r[6]=(_Float16)b.z; r[7]=(_Float16)b.w;
  *(v8h*)(dst + i) = r;
}

// ---- Kernel 1: Q/K/V projection (x @ W^T) + fused axial RoPE ----
// grid = (M/16, 3), block = 256. Wave w computes rows m0..m0+15, cols w*64..w*64+63
// (one head per wave); A fragment reused across 4 WMMAs per k-step.
__global__ __launch_bounds__(256)
void qkv_rope_kernel(const _Float16* __restrict__ xf,   // [M,512] f16
                     const _Float16* __restrict__ wf,   // [3][512][512] f16 (q,k,v)
                     _Float16* __restrict__ q_ws,
                     _Float16* __restrict__ k_ws,
                     _Float16* __restrict__ vt_ws) {
  __shared__ __align__(16) _Float16 xl[16][520];
  const int m0    = blockIdx.x * 16;
  const int which = blockIdx.y;
  const int t     = threadIdx.x;

  { // stage x tile (16 x 512 f16) -> LDS
    int row = t >> 4;
    int cb  = (t & 15) * 32;
    const _Float16* src = xf + (size_t)(m0 + row) * D_EMB + cb;
    #pragma unroll
    for (int u = 0; u < 32; u += 8)
      *(v8h*)(&xl[row][cb + u]) = *(const v8h*)(src + u);
  }
  __syncthreads();

  const int lane = t & 31;
  const int wave = t >> 5;
  const int n0   = wave * 64;                       // head h == wave
  const _Float16* W = wf + (size_t)which * WELEM;

  v8f acc[4] = {v8f{}, v8f{}, v8f{}, v8f{}};
  for (int k = 0; k < D_EMB; k += 32) {
    v16h a = load_frag_a_h(&xl[0][k], 520, lane);
    #pragma unroll
    for (int nt = 0; nt < 4; ++nt) {
      v16h b = load_frag_b_h(W + (size_t)(n0 + nt * 16) * D_EMB + k, D_EMB, lane);
      acc[nt] = wmma_f16(a, b, acc[nt]);
    }
  }

  const int col   = lane & 15;
  const int rhalf = (lane >> 4) << 3;

  if (which == 2) {
    // V: no rope; store transposed [B,H,hd,S] for contiguous B-frags in P·V
    #pragma unroll
    for (int nt = 0; nt < 4; ++nt) {
      int n = n0 + nt * 16 + col;
      int h = n >> 6, d = n & 63;
      #pragma unroll
      for (int i = 0; i < 8; ++i) {
        int m  = m0 + i + rhalf;
        int bb = m >> 12;
        int s  = m & (SEQ - 1);
        vt_ws[(((size_t)bb * NH + h) * HD + d) * SEQ + s] = (_Float16)acc[nt][i];
      }
    }
  } else {
    // axial RoPE: dims 0..31 use t_idx = s/32, dims 32..63 use v_idx = s%32
    _Float16* dst = (which == 0) ? q_ws : k_ws;
    const float oscale = (which == 0) ? 0.125f : 1.0f;  // fold hd^-0.5 into Q
    #pragma unroll
    for (int nt = 0; nt < 4; ++nt) {
      int n = n0 + nt * 16 + col;
      int h = n >> 6, d = n & 63;
      int   j    = (d & 31) >> 1;
      float invf = __powf(10000.0f, -(float)j * (1.0f / 16.0f));
      float sgn  = (d & 1) ? 1.0f : -1.0f;  // even lane holds xe: -sin; odd: +sin
      bool  blockv = (d >= 32);
      #pragma unroll
      for (int i = 0; i < 8; ++i) {
        float xv      = acc[nt][i];
        float partner = __shfl_xor(xv, 1);  // rope pair sits in adjacent lane
        int m  = m0 + i + rhalf;
        int bb = m >> 12;
        int s  = m & (SEQ - 1);
        int idx = blockv ? (s & 31) : (s >> 5);
        float ang = (float)idx * invf;
        float cs, sn;
        __sincosf(ang, &cs, &sn);
        float outv = (xv * cs + sgn * sn * partner) * oscale;
        dst[(((size_t)bb * NH + h) * SEQ + s) * HD + d] = (_Float16)outv;
      }
    }
  }
}

// ---- Kernel 2: flash attention, one (b,h,128-q-row) tile per block ----
// grid = (S/128, H, B), block = 256 (8 waves x 16 q rows)
__global__ __launch_bounds__(256)
void flash_attn_kernel(const _Float16* __restrict__ q_ws,
                       const _Float16* __restrict__ k_ws,
                       const _Float16* __restrict__ vt_ws,
                       _Float16* __restrict__ o_ws) {
  __shared__ __align__(16) _Float16 Kl[64][72];     // [key j][d]
  __shared__ __align__(16) _Float16 Vl[64][72];     // [d][key j] (V transposed)
  __shared__ __align__(16) _Float16 Pl[8][16][72];  // per-wave P round-trip

  const int t    = threadIdx.x;
  const int lane = t & 31;
  const int wave = t >> 5;
  const int h    = blockIdx.y;
  const int b    = blockIdx.z;
  const int bh   = b * NH + h;
  const int q0   = blockIdx.x * 128 + wave * 16;
  const int rhalf = (lane >> 4) << 3;
  const int cb    = lane & 15;

  const _Float16* qbase = q_ws + ((size_t)bh * SEQ + q0) * HD;
  v16h qf0 = load_frag_a_h(qbase + 0,  HD, lane);
  v16h qf1 = load_frag_a_h(qbase + 32, HD, lane);

  float mrow[8], lrow[8], oacc[4][8];
  #pragma unroll
  for (int i = 0; i < 8; ++i) { mrow[i] = -1e30f; lrow[i] = 0.0f; }
  #pragma unroll
  for (int dt = 0; dt < 4; ++dt)
    #pragma unroll
    for (int i = 0; i < 8; ++i) oacc[dt][i] = 0.0f;

  const int ldrow = t >> 2;          // 0..63
  const int ldcol = (t & 3) * 16;    // 0,16,32,48
  const _Float16* ksrc0 = k_ws  + (size_t)bh * SEQ * HD + (size_t)ldrow * HD + ldcol;
  const _Float16* vsrc0 = vt_ws + ((size_t)bh * HD + ldrow) * SEQ + ldcol;

  for (int kv0 = 0; kv0 < SEQ; kv0 += 64) {
    __syncthreads();  // all waves done with previous K/V tile
    {
      const _Float16* ks = ksrc0 + (size_t)kv0 * HD;
      *(v8h*)(&Kl[ldrow][ldcol])     = *(const v8h*)(ks);
      *(v8h*)(&Kl[ldrow][ldcol + 8]) = *(const v8h*)(ks + 8);
      const _Float16* vs = vsrc0 + kv0;
      *(v8h*)(&Vl[ldrow][ldcol])     = *(const v8h*)(vs);
      *(v8h*)(&Vl[ldrow][ldcol + 8]) = *(const v8h*)(vs + 8);
    }
    __syncthreads();

    // S = Q * K^T (scale pre-folded into Q)
    v8f sf[4];
    #pragma unroll
    for (int jt = 0; jt < 4; ++jt) {
      v8f s = {};
      v16h b0 = load_frag_b_h(&Kl[jt * 16][0], 72, lane);
      s = wmma_f16(qf0, b0, s);
      v16h b1 = load_frag_b_h(&Kl[jt * 16][32], 72, lane);
      s = wmma_f16(qf1, b1, s);
      sf[jt] = s;
    }

    // online softmax: 16-lane butterflies (xor<16 stays inside the lane half)
    float alpha_a[8];
    #pragma unroll
    for (int i = 0; i < 8; ++i) {
      float t0 = fmaxf(fmaxf(sf[0][i], sf[1][i]), fmaxf(sf[2][i], sf[3][i]));
      t0 = fmaxf(t0, __shfl_xor(t0, 1));
      t0 = fmaxf(t0, __shfl_xor(t0, 2));
      t0 = fmaxf(t0, __shfl_xor(t0, 4));
      t0 = fmaxf(t0, __shfl_xor(t0, 8));
      float mn    = fmaxf(mrow[i], t0);
      float alpha = __expf(mrow[i] - mn);
      mrow[i]    = mn;
      alpha_a[i] = alpha;
      float p0 = __expf(sf[0][i] - mn);
      float p1 = __expf(sf[1][i] - mn);
      float p2 = __expf(sf[2][i] - mn);
      float p3 = __expf(sf[3][i] - mn);
      float rs = p0 + p1 + p2 + p3;
      rs += __shfl_xor(rs, 1);
      rs += __shfl_xor(rs, 2);
      rs += __shfl_xor(rs, 4);
      rs += __shfl_xor(rs, 8);
      lrow[i] = lrow[i] * alpha + rs;
      int r = i + rhalf;
      Pl[wave][r][0  + cb] = (_Float16)p0;   // C-layout -> LDS -> A-layout
      Pl[wave][r][16 + cb] = (_Float16)p1;
      Pl[wave][r][32 + cb] = (_Float16)p2;
      Pl[wave][r][48 + cb] = (_Float16)p3;
    }
    #pragma unroll
    for (int dt = 0; dt < 4; ++dt)
      #pragma unroll
      for (int i = 0; i < 8; ++i) oacc[dt][i] *= alpha_a[i];

    __syncthreads();  // order P store -> P load

    // O += P * V
    v16h pa0 = load_frag_a_h(&Pl[wave][0][0],  72, lane);
    v16h pa1 = load_frag_a_h(&Pl[wave][0][32], 72, lane);
    #pragma unroll
    for (int dt = 0; dt < 4; ++dt) {
      v8f o;
      #pragma unroll
      for (int i = 0; i < 8; ++i) o[i] = oacc[dt][i];
      v16h vb0 = load_frag_b_h(&Vl[dt * 16][0], 72, lane);
      o = wmma_f16(pa0, vb0, o);
      v16h vb1 = load_frag_b_h(&Vl[dt * 16][32], 72, lane);
      o = wmma_f16(pa1, vb1, o);
      #pragma unroll
      for (int i = 0; i < 8; ++i) oacc[dt][i] = o[i];
    }
  }

  // normalize, write O in [B,S,D] f16 for the output projection
  #pragma unroll
  for (int dt = 0; dt < 4; ++dt) {
    int n = h * HD + dt * 16 + cb;
    #pragma unroll
    for (int i = 0; i < 8; ++i) {
      int s = q0 + i + rhalf;
      float val = oacc[dt][i] / lrow[i];
      o_ws[((size_t)b * SEQ + s) * D_EMB + n] = (_Float16)val;
    }
  }
}

// ---- Kernel 3: output projection y = O @ Wo^T (f32 out) ----
// grid = (M/16), block = 256; wave w covers cols w*64..w*64+63
__global__ __launch_bounds__(256)
void out_proj_kernel(const _Float16* __restrict__ o_ws,
                     const _Float16* __restrict__ wo16,
                     float* __restrict__ y) {
  __shared__ __align__(16) _Float16 al[16][520];
  const int m0 = blockIdx.x * 16;
  const int t  = threadIdx.x;
  {
    int row = t >> 4;
    int cb  = (t & 15) * 32;
    const _Float16* src = o_ws + (size_t)(m0 + row) * D_EMB + cb;
    #pragma unroll
    for (int u = 0; u < 32; u += 8)
      *(v8h*)(&al[row][cb + u]) = *(const v8h*)(src + u);
  }
  __syncthreads();

  const int lane = t & 31;
  const int wave = t >> 5;
  const int n0   = wave * 64;

  v8f acc[4] = {v8f{}, v8f{}, v8f{}, v8f{}};
  for (int k = 0; k < D_EMB; k += 32) {
    v16h a = load_frag_a_h(&al[0][k], 520, lane);
    #pragma unroll
    for (int nt = 0; nt < 4; ++nt) {
      v16h b = load_frag_b_h(wo16 + (size_t)(n0 + nt * 16) * D_EMB + k, D_EMB, lane);
      acc[nt] = wmma_f16(a, b, acc[nt]);
    }
  }

  const int cbl   = lane & 15;
  const int rhalf = (lane >> 4) << 3;
  #pragma unroll
  for (int nt = 0; nt < 4; ++nt) {
    #pragma unroll
    for (int i = 0; i < 8; ++i) {
      int m = m0 + i + rhalf;
      y[(size_t)m * D_EMB + n0 + nt * 16 + cbl] = acc[nt][i];
    }
  }
}

extern "C" void kernel_launch(void* const* d_in, const int* in_sizes, int n_in,
                              void* d_out, int out_size, void* d_ws, size_t ws_size,
                              hipStream_t stream) {
  (void)in_sizes; (void)n_in; (void)out_size; (void)ws_size;
  const float* x  = (const float*)d_in[0];
  const float* Wq = (const float*)d_in[1];
  const float* Wk = (const float*)d_in[2];
  const float* Wv = (const float*)d_in[3];
  const float* Wo = (const float*)d_in[4];
  float* y = (float*)d_out;

  _Float16* ws = (_Float16*)d_ws;
  _Float16* wqkv16 = ws;                     // [3][512][512] f16
  _Float16* wo16   = wqkv16 + 3 * WELEM;     // [512][512] f16
  _Float16* xf16   = wo16 + WELEM;           // [B*S, 512] f16
  _Float16* q_ws   = xf16 + XELEM;           // [B,H,S,hd] roped, *scale
  _Float16* k_ws   = q_ws + XELEM;           // [B,H,S,hd] roped
  _Float16* vt_ws  = k_ws + XELEM;           // [B,H,hd,S] transposed
  _Float16* o_ws   = vt_ws + XELEM;          // [B,S,D]

  // one-time f32 -> f16 pre-pass (weights + x); 8 elems/thread
  cvt_f32_f16_kernel<<<WELEM / 2048, 256, 0, stream>>>(Wq, wqkv16);
  cvt_f32_f16_kernel<<<WELEM / 2048, 256, 0, stream>>>(Wk, wqkv16 + WELEM);
  cvt_f32_f16_kernel<<<WELEM / 2048, 256, 0, stream>>>(Wv, wqkv16 + 2 * WELEM);
  cvt_f32_f16_kernel<<<WELEM / 2048, 256, 0, stream>>>(Wo, wo16);
  cvt_f32_f16_kernel<<<(unsigned)(XELEM / 2048), 256, 0, stream>>>(x, xf16);

  dim3 g1((NB * SEQ) / 16, 3);
  qkv_rope_kernel<<<g1, 256, 0, stream>>>(xf16, wqkv16, q_ws, k_ws, vt_ws);

  dim3 g2(SEQ / 128, NH, NB);
  flash_attn_kernel<<<g2, 256, 0, stream>>>(q_ws, k_ws, vt_ws, o_ws);

  out_proj_kernel<<<(NB * SEQ) / 16, 256, 0, stream>>>(o_ws, wo16, y);
}
